// CRModule_39676907888270
// MI455X (gfx1250) — matrix operand order repaired and checked by reading, not compiled, not run on hardware
//
#include <hip/hip_runtime.h>
#include <hip/hip_bf16.h>

typedef __attribute__((ext_vector_type(16))) __bf16 v16bf;
typedef __attribute__((ext_vector_type(8)))  __bf16 v8bf;
typedef __attribute__((ext_vector_type(8)))  float  v8f;

constexpr int T  = 4096;   // reduction dim (B*N rows of xf)
constexpr int C  = 4096;   // feature dim of x / fc_weight
constexpr int HC = 2048;   // C/2 (output is HC x HC)

// ---------------------------------------------------------------- zero ws
__global__ void zero_f32(float* p, int n) {
  int i = blockIdx.x * blockDim.x + threadIdx.x;
  if (i < n) p[i] = 0.0f;
}

// ------------------------------------------------ column sums / sum of squares
// out[col] += sum over this block's row chunk of in[r][col] (or its square).
template <bool SQ>
__global__ void colreduce(const float* __restrict__ in, int rows_per_block,
                          int cols, float* __restrict__ out) {
  const int col = blockIdx.x * blockDim.x + threadIdx.x;
  const float* p = in + (size_t)blockIdx.y * rows_per_block * cols + col;
  float acc = 0.0f;
  for (int r = 0; r < rows_per_block; ++r) {
    float v = *p;
    acc += SQ ? v * v : v;
    p += cols;
  }
  atomicAdd(&out[col], acc);
}

// ------------------------------------- split even/odd columns, transpose, cvt bf16
// xaT[i][t] = x[t][2i], xbT[j][t] = x[t][2j+1]   (both HC x T, K contiguous)
__global__ __launch_bounds__(256) void split_transpose_bf16(
    const float* __restrict__ x, __bf16* __restrict__ xaT,
    __bf16* __restrict__ xbT) {
  __shared__ __bf16 tile[64][72];  // [c_local][t_local], padded, 16B-aligned rows
  const int c0 = blockIdx.x * 64;
  const int t0 = blockIdx.y * 64;
  const int tid = threadIdx.x;
  const int tl = tid >> 4;          // 0..15
  const int c4 = (tid & 15) * 4;    // 0..60
#pragma unroll
  for (int p = 0; p < 4; ++p) {
    const int t = tl + p * 16;
    const float4 v = *(const float4*)(x + (size_t)(t0 + t) * C + c0 + c4);
    tile[c4 + 0][t] = (__bf16)v.x;
    tile[c4 + 1][t] = (__bf16)v.y;
    tile[c4 + 2][t] = (__bf16)v.z;
    tile[c4 + 3][t] = (__bf16)v.w;
  }
  __syncthreads();
  // 64 output rows x 64 bf16 = 512 chunks of 8 bf16 (16B); 2 per thread
#pragma unroll
  for (int q = 0; q < 2; ++q) {
    const int h  = tid + q * 256;
    const int cl = h >> 3;
    const int t8 = (h & 7) * 8;
    const int c  = c0 + cl;
    __bf16* dst = (c & 1) ? (xbT + (size_t)(c >> 1) * T + t0 + t8)
                          : (xaT + (size_t)(c >> 1) * T + t0 + t8);
    *(v8bf*)dst = *(const v8bf*)&tile[cl][t8];
  }
}

// ---------------------------------------------------------------- main GEMM
// scores[i][j] = (colsum[2i]+colsum[2j+1])^2 *
//               max(colsq[2i]+colsq[2j+1] - 2*sum_t xaT[i][t]*xbT[j][t], 0)
__global__ __launch_bounds__(256) void gemm_scores(
    const __bf16* __restrict__ xaT, const __bf16* __restrict__ xbT,
    const float* __restrict__ colsum, const float* __restrict__ colsq,
    float* __restrict__ out) {
  constexpr int LDT = 72;                 // LDS row stride (bf16): 144B, 16B aligned
  __shared__ __bf16 As[128 * LDT];
  __shared__ __bf16 Bs[128 * LDT];

  const int i0 = blockIdx.y * 128;
  const int j0 = blockIdx.x * 128;
  const int tid  = threadIdx.x;
  const int wave = tid >> 5;              // 8 waves (wave32)
  const int lane = tid & 31;
  const int h    = lane >> 4;             // half-wave select
  const int l16  = lane & 15;
  const int wm = (wave & 1) * 64;         // wave tile: 64(M) x 32(N)
  const int wn = (wave >> 1) * 32;

  v8f acc[4][2];
#pragma unroll
  for (int a = 0; a < 4; ++a)
#pragma unroll
    for (int b = 0; b < 2; ++b)
#pragma unroll
      for (int e = 0; e < 8; ++e) acc[a][b][e] = 0.0f;

  for (int kt = 0; kt < T; kt += 64) {
    // global -> regs (issued before the barrier: overlaps previous tile's math)
    uint4 ra[4], rb[4];
#pragma unroll
    for (int q = 0; q < 4; ++q) {
      const int ch  = tid + q * 256;      // 0..1023 chunks of 8 bf16
      const int row = ch >> 3;            // 0..127
      const int c8  = (ch & 7) * 8;       // 0..56
      ra[q] = *(const uint4*)(xaT + (size_t)(i0 + row) * T + kt + c8);
      rb[q] = *(const uint4*)(xbT + (size_t)(j0 + row) * T + kt + c8);
    }
    __syncthreads();                      // previous compute done with LDS
#pragma unroll
    for (int q = 0; q < 4; ++q) {
      const int ch  = tid + q * 256;
      const int row = ch >> 3;
      const int c8  = (ch & 7) * 8;
      *(uint4*)&As[row * LDT + c8] = ra[q];
      *(uint4*)&Bs[row * LDT + c8] = rb[q];
    }
    __syncthreads();

#pragma unroll
    for (int ks = 0; ks < 2; ++ks) {      // two K=32 WMMA steps per stage
      const int ko = ks * 32;
      v16bf afrag[4];
      v16bf bfrag[2];
#pragma unroll
      for (int fm = 0; fm < 4; ++fm) {
        // A 16x32 bf16 layout: lanes0-15 K=0..7 & 16..23, lanes16-31 K=8..15 & 24..31
        const __bf16* base = &As[(wm + fm * 16 + l16) * LDT + ko + 8 * h];
        const v8bf lo = *(const v8bf*)base;
        const v8bf hi = *(const v8bf*)(base + 16);
#pragma unroll
        for (int e = 0; e < 8; ++e) { afrag[fm][e] = lo[e]; afrag[fm][e + 8] = hi[e]; }
      }
#pragma unroll
      for (int fn = 0; fn < 2; ++fn) {
        // B 32x16 bf16 layout: lanes0-15 K=0..7 & 16..23, lanes16-31 K=8..15 & 24..31
        const __bf16* base = &Bs[(wn + fn * 16 + l16) * LDT + ko + 8 * h];
        const v8bf lo = *(const v8bf*)base;
        const v8bf hi = *(const v8bf*)(base + 16);
#pragma unroll
        for (int e = 0; e < 8; ++e) { bfrag[fn][e] = lo[e]; bfrag[fn][e + 8] = hi[e]; }
      }
#pragma unroll
      for (int fm = 0; fm < 4; ++fm)
#pragma unroll
        for (int fn = 0; fn < 2; ++fn)
          acc[fm][fn] = __builtin_amdgcn_wmma_f32_16x16x32_bf16(
              false, afrag[fm], false, bfrag[fn], (short)0, acc[fm][fn],
              false, false);
    }
  }

  // ---- fused epilogue: sqrt/square cancel -> w^2 * max(sq, 0)
#pragma unroll
  for (int fm = 0; fm < 4; ++fm) {
#pragma unroll
    for (int fn = 0; fn < 2; ++fn) {
      const int n   = j0 + wn + fn * 16 + l16;
      const float wb  = colsum[2 * n + 1];
      const float rbn = colsq[2 * n + 1];
#pragma unroll
      for (int r = 0; r < 8; ++r) {
        const int m = i0 + wm + fm * 16 + r + 8 * h;   // C/D: VGPR r -> M = r + 8*half
        const float wa  = colsum[2 * m];
        const float ram = colsq[2 * m];
        float sq = ram + rbn - 2.0f * acc[fm][fn][r];
        sq = sq > 0.0f ? sq : 0.0f;
        const float w = wa + wb;
        out[(size_t)m * HC + n] = w * w * sq;
      }
    }
  }
}

// ---------------------------------------------------------------- launcher
extern "C" void kernel_launch(void* const* d_in, const int* in_sizes, int n_in,
                              void* d_out, int out_size, void* d_ws, size_t ws_size,
                              hipStream_t stream) {
  const float* x  = (const float*)d_in[0];   // (2,2048,4096) fp32
  const float* fw = (const float*)d_in[1];   // (12288,4096) fp32
  float* out = (float*)d_out;                // (2048,2048) fp32

  char* ws = (char*)d_ws;
  float*  colsum = (float*)ws;                             // 4096 f32
  float*  colsq  = (float*)(ws + 16384);                   // 4096 f32
  __bf16* xaT    = (__bf16*)(ws + 32768);                  // 2048*4096 bf16
  __bf16* xbT    = (__bf16*)(ws + 32768 + (size_t)HC * T * 2);

  // colsum & colsq are contiguous: one zero pass covers both (ws is poisoned)
  zero_f32<<<dim3(32), 256, 0, stream>>>(colsum, 8192);
  // weight column sums: 12288 rows = 96 chunks of 128
  colreduce<false><<<dim3(16, 96), 256, 0, stream>>>(fw, 128, C, colsum);
  // x column sums of squares: 4096 rows = 64 chunks of 64
  colreduce<true><<<dim3(16, 64), 256, 0, stream>>>(x, 64, C, colsq);
  // split even/odd + transpose + bf16
  split_transpose_bf16<<<dim3(64, 64), 256, 0, stream>>>(x, xaT, xbT);
  // 2048x2048 output, 128x128 tiles
  gemm_scores<<<dim3(16, 16), 256, 0, stream>>>(xaT, xbT, colsum, colsq, out);
}
